// Discriminator_4097398800565
// MI455X (gfx1250) — matrix operand order repaired and checked by reading, not compiled, not run on hardware
//
#include <hip/hip_runtime.h>
#include <hip/hip_bf16.h>
#include <math.h>

typedef __bf16 v16bf __attribute__((ext_vector_type(16)));
typedef float  v8f   __attribute__((ext_vector_type(8)));

#define B_ 16
#define N_ 128
#define M_ 64
#define D_ 512
#define H_ 4
#define K_ 128
#define L_ 3

#define WMMA_BF16(A, Bf, C) \
  __builtin_amdgcn_wmma_f32_16x16x32_bf16(false, (A), false, (Bf), (short)0, (C), false, false)

__device__ __forceinline__ __bf16 f2bf(float f) {
  union { float f; unsigned u; } v; v.f = f;
  unsigned r = v.u + 0x7FFFu + ((v.u >> 16) & 1u);   // round-to-nearest-even
  unsigned short s = (unsigned short)(r >> 16);
  __bf16 o; __builtin_memcpy(&o, &s, 2); return o;
}

// ---------------------------------------------------------------------------
// WMMA B-fragment layout for a 512x512 bf16 matrix:
//   Wb[((kb*32 + ct)*32 + lane)*16 + e]   (32 B contiguous per lane)
//   col = ct*16 + lane%16,  d = kb*32 + 16*(lane/16) + e   (contiguous k order, same as A)
// ---------------------------------------------------------------------------

// Pack lin_w [H,D,K] -> head-concatenated [D, H*K] fragments; zero head acc.
__global__ void pack_lin_kernel(const float* __restrict__ lin_w,
                                unsigned short* __restrict__ Wb,
                                float* __restrict__ head_sum) {
  int idx = blockIdx.x * blockDim.x + threadIdx.x;
  if (idx < B_ * D_) head_sum[idx] = 0.0f;
  if (idx < 16 * 32 * 32 * 16) {
    int e    = idx & 15;
    int lane = (idx >> 4) & 31;
    int ctg  = (idx >> 9) & 31;
    int kb   = idx >> 14;
    int col  = ctg * 16 + (lane & 15);
    int h    = col >> 7;
    int k    = col & (K_ - 1);
    int d    = kb * 32 + ((lane >> 4) << 4) + e;
    union { __bf16 b; unsigned short s; } u;
    u.b = f2bf(lin_w[(h * D_ + d) * K_ + k]);
    Wb[idx] = u.s;
  }
}

// Pack a row-major [D,D] matrix into the same fragment layout.
__global__ void pack_dd_kernel(const float* __restrict__ W,
                               unsigned short* __restrict__ Wb) {
  int idx = blockIdx.x * blockDim.x + threadIdx.x;
  if (idx < 16 * 32 * 32 * 16) {
    int e    = idx & 15;
    int lane = (idx >> 4) & 31;
    int ctg  = (idx >> 9) & 31;
    int kb   = idx >> 14;
    int col  = ctg * 16 + (lane & 15);
    int d    = kb * 32 + ((lane >> 4) << 4) + e;
    union { __bf16 b; unsigned short s; } u;
    u.b = f2bf(W[d * D_ + col]);
    Wb[idx] = u.s;
  }
}

// Store an f32 element into the WMMA A-fragment LDS image (16x32 tiles):
// lanes 0-15 row=lane hold K 0-15, lanes 16-31 hold K 16-31.
__device__ __forceinline__ void store_afrag(__bf16* As, int m, int d, float v) {
  int rt = m >> 4, rrow = m & 15;
  int kb = d >> 5, kd = d & 31;
  int lane, e;
  lane = rrow + ((kd >> 4) << 4); e = kd & 15;
  As[((rt * 16 + kb) * 32 + lane) * 16 + e] = f2bf(v);
}

// Write one 16x16 f32 C tile (8 values / lane) + bias to a [*,512] f32 image.
__device__ __forceinline__ void store_ctile(float* __restrict__ dst, v8f c,
                                            int row0, int col, float lb) {
#pragma unroll
  for (int v = 0; v < 8; ++v) dst[(row0 + v) * D_ + col] = c[v] + lb;
}

// ---------------------------------------------------------------------------
// Register-tiled 64x512x512 GEMM body (per wave): 4x4 tile of accumulators
// kept live across the single kb loop; every load is kb-dependent so nothing
// is hoistable (avoids the LICM -> scratch-spill failure of the ctl-outer
// form). 16 v_wmma per iteration sharing 4 B fragments.
// ---------------------------------------------------------------------------
#define GEMM_64x512(ASV, WBV, LANE, CB)                                        \
  v8f c00 = {}, c10 = {}, c20 = {}, c30 = {};                                  \
  v8f c01 = {}, c11 = {}, c21 = {}, c31 = {};                                  \
  v8f c02 = {}, c12 = {}, c22 = {}, c32 = {};                                  \
  v8f c03 = {}, c13 = {}, c23 = {}, c33 = {};                                  \
  _Pragma("unroll 1")                                                          \
  for (int kb = 0; kb < 16; ++kb) {                                            \
    const v16bf a0 = (ASV)[(0 * 16 + kb) * 32 + (LANE)];                       \
    const v16bf a1 = (ASV)[(1 * 16 + kb) * 32 + (LANE)];                       \
    const v16bf a2 = (ASV)[(2 * 16 + kb) * 32 + (LANE)];                       \
    const v16bf a3 = (ASV)[(3 * 16 + kb) * 32 + (LANE)];                       \
    const v16bf b0 = (WBV)[(kb * 32 + (CB) + 0) * 32 + (LANE)];                \
    c00 = WMMA_BF16(a0, b0, c00);  c10 = WMMA_BF16(a1, b0, c10);               \
    c20 = WMMA_BF16(a2, b0, c20);  c30 = WMMA_BF16(a3, b0, c30);               \
    const v16bf b1 = (WBV)[(kb * 32 + (CB) + 1) * 32 + (LANE)];                \
    c01 = WMMA_BF16(a0, b1, c01);  c11 = WMMA_BF16(a1, b1, c11);               \
    c21 = WMMA_BF16(a2, b1, c21);  c31 = WMMA_BF16(a3, b1, c31);               \
    const v16bf b2 = (WBV)[(kb * 32 + (CB) + 2) * 32 + (LANE)];                \
    c02 = WMMA_BF16(a0, b2, c02);  c12 = WMMA_BF16(a1, b2, c12);               \
    c22 = WMMA_BF16(a2, b2, c22);  c32 = WMMA_BF16(a3, b2, c32);               \
    const v16bf b3 = (WBV)[(kb * 32 + (CB) + 3) * 32 + (LANE)];                \
    c03 = WMMA_BF16(a0, b3, c03);  c13 = WMMA_BF16(a1, b3, c13);               \
    c23 = WMMA_BF16(a2, b3, c23);  c33 = WMMA_BF16(a3, b3, c33);               \
  }

// ---------------------------------------------------------------------------
// 64-row WMMA GEMM: y[row0+0..63, :] = x[row0+0..63, :] @ W + bias
// ---------------------------------------------------------------------------
__global__ void __launch_bounds__(256, 1)
gemm64_kernel(const float* __restrict__ x,
              const unsigned short* __restrict__ Wb,
              const float* __restrict__ bias,
              float* __restrict__ y) {
  extern __shared__ char smem[];
  __bf16* As = (__bf16*)smem;                 // 64 KB fragment image
  const int tid  = threadIdx.x;
  const int row0 = blockIdx.x * 64;

  for (int idx = tid; idx < 64 * D_; idx += 256) {
    int m = idx >> 9, d = idx & (D_ - 1);
    store_afrag(As, m, d, x[(row0 + m) * D_ + d]);
  }
  __syncthreads();

  const int wave = tid >> 5;
  const int lane = tid & 31;
  const v16bf* Asv = (const v16bf*)As;
  const v16bf* Wbv = (const v16bf*)Wb;
  const int cb = wave * 4;

  GEMM_64x512(Asv, Wbv, lane, cb)

  const int col0 = cb * 16 + (lane & 15);
  const int rowb = row0 + ((lane >> 4) << 3);       // C: row = v + 8*(lane/16)
  const float lb0 = bias[col0];
  const float lb1 = bias[col0 + 16];
  const float lb2 = bias[col0 + 32];
  const float lb3 = bias[col0 + 48];
  store_ctile(y, c00, rowb +  0, col0,      lb0);
  store_ctile(y, c10, rowb + 16, col0,      lb0);
  store_ctile(y, c20, rowb + 32, col0,      lb0);
  store_ctile(y, c30, rowb + 48, col0,      lb0);
  store_ctile(y, c01, rowb +  0, col0 + 16, lb1);
  store_ctile(y, c11, rowb + 16, col0 + 16, lb1);
  store_ctile(y, c21, rowb + 32, col0 + 16, lb1);
  store_ctile(y, c31, rowb + 48, col0 + 16, lb1);
  store_ctile(y, c02, rowb +  0, col0 + 32, lb2);
  store_ctile(y, c12, rowb + 16, col0 + 32, lb2);
  store_ctile(y, c22, rowb + 32, col0 + 32, lb2);
  store_ctile(y, c32, rowb + 48, col0 + 32, lb2);
  store_ctile(y, c03, rowb +  0, col0 + 48, lb3);
  store_ctile(y, c13, rowb + 16, col0 + 48, lb3);
  store_ctile(y, c23, rowb + 32, col0 + 48, lb3);
  store_ctile(y, c33, rowb + 48, col0 + 48, lb3);
}

// ---------------------------------------------------------------------------
// Main fused kernel: one block per (b,n).
//   A[m,d] = relu(w[b,n,d] * r[b,m,d])  (bf16, WMMA A-fragment layout in LDS)
//   t[64,512] = A @ W + lin_b           (v_wmma_f32_16x16x32_bf16)
//   s = t . score_w + score_b ; softmax over m ; * mask
//   head += (1/N) * sum_m wts[m,h] * t[m,:]   (atomic into global)
// ---------------------------------------------------------------------------
__global__ void __launch_bounds__(256, 1)
main_kernel(const float* __restrict__ wbuf,          // [B,N,D]
            const float* __restrict__ rbuf,          // [B,M,D]
            const float* __restrict__ mask,          // [B,M]
            const unsigned short* __restrict__ Wb,   // bf16 fragments
            const float* __restrict__ lin_b,         // [H*K]
            const float* __restrict__ score_w,       // [H,K]
            const float* __restrict__ score_b,       // [H]
            float* __restrict__ head_sum)            // [B,D]
{
  extern __shared__ char smem[];
  __bf16* As = (__bf16*)smem;                              // 64 KB
  float*  ts = (float*)(smem + 65536);                     // 128 KB  [64][512]
  float*  sc = (float*)(smem + 65536 + 131072);            // 1 KB    [64][4]
  float*  wv = (float*)(smem + 65536 + 131072 + 1024);     // 2 KB    [512]

  const int tid = threadIdx.x;
  const int b = blockIdx.x / N_;
  const int n = blockIdx.x % N_;

  for (int i = tid; i < D_; i += 256) wv[i] = wbuf[(b * N_ + n) * D_ + i];
  __syncthreads();

  for (int idx = tid; idx < M_ * D_; idx += 256) {
    int m = idx >> 9, d = idx & (D_ - 1);
    float v = fmaxf(wv[d] * rbuf[(b * M_ + m) * D_ + d], 0.0f);
    store_afrag(As, m, d, v);
  }
  __syncthreads();

  const int wave = tid >> 5;
  const int lane = tid & 31;
  const v16bf* Asv = (const v16bf*)As;
  const v16bf* Wbv = (const v16bf*)Wb;
  const int cb = wave * 4;

  GEMM_64x512(Asv, Wbv, lane, cb)

  const int col0 = cb * 16 + (lane & 15);
  const int rowb = (lane >> 4) << 3;
  const float lb0 = lin_b[col0];
  const float lb1 = lin_b[col0 + 16];
  const float lb2 = lin_b[col0 + 32];
  const float lb3 = lin_b[col0 + 48];
  store_ctile(ts, c00, rowb +  0, col0,      lb0);
  store_ctile(ts, c10, rowb + 16, col0,      lb0);
  store_ctile(ts, c20, rowb + 32, col0,      lb0);
  store_ctile(ts, c30, rowb + 48, col0,      lb0);
  store_ctile(ts, c01, rowb +  0, col0 + 16, lb1);
  store_ctile(ts, c11, rowb + 16, col0 + 16, lb1);
  store_ctile(ts, c21, rowb + 32, col0 + 16, lb1);
  store_ctile(ts, c31, rowb + 48, col0 + 16, lb1);
  store_ctile(ts, c02, rowb +  0, col0 + 32, lb2);
  store_ctile(ts, c12, rowb + 16, col0 + 32, lb2);
  store_ctile(ts, c22, rowb + 32, col0 + 32, lb2);
  store_ctile(ts, c32, rowb + 48, col0 + 32, lb2);
  store_ctile(ts, c03, rowb +  0, col0 + 48, lb3);
  store_ctile(ts, c13, rowb + 16, col0 + 48, lb3);
  store_ctile(ts, c23, rowb + 32, col0 + 48, lb3);
  store_ctile(ts, c33, rowb + 48, col0 + 48, lb3);
  __syncthreads();

  // Scores: one (m,h) pair per thread (exactly 256 pairs).
  {
    const int m = tid >> 2, h = tid & 3;
    float s = score_b[h];
    const float* tr = &ts[m * D_ + h * K_];
    const float* sw = &score_w[h * K_];
    for (int k = 0; k < K_; ++k) s = fmaf(tr[k], sw[k], s);
    sc[m * 4 + h] = s;
  }
  __syncthreads();

  // Softmax over m per head, then multiply by mask (post-softmax, as in ref).
  if (tid < H_) {
    const int h = tid;
    float mx = -3.4e38f;
    for (int m = 0; m < M_; ++m) mx = fmaxf(mx, sc[m * 4 + h]);
    float sum = 0.0f;
    for (int m = 0; m < M_; ++m) sum += __expf(sc[m * 4 + h] - mx);
    const float inv = 1.0f / sum;
    for (int m = 0; m < M_; ++m)
      sc[m * 4 + h] = __expf(sc[m * 4 + h] - mx) * inv * mask[b * M_ + m];
  }
  __syncthreads();

  // head[b, c] += (1/N) * sum_m wts[m, h(c)] * t[m, c]
  for (int c = tid; c < D_; c += 256) {
    const int h = c >> 7;
    float a = 0.0f;
    for (int m = 0; m < M_; ++m) a = fmaf(sc[m * 4 + h], ts[m * D_ + c], a);
    atomicAdd(&head_sum[b * D_ + c], a * (1.0f / N_));
  }
}

// ---------------------------------------------------------------------------
// Final MLP: out = relu(head @ final_w + final_b) @ fc_w + fc_b -> [B,L]
// ---------------------------------------------------------------------------
__global__ void final_kernel(const float* __restrict__ head_sum,
                             const float* __restrict__ final_w,
                             const float* __restrict__ final_b,
                             const float* __restrict__ fc_w,
                             const float* __restrict__ fc_b,
                             float* __restrict__ out) {
  __shared__ float hv[D_];
  __shared__ float ov[D_];
  const int b = blockIdx.x, tid = threadIdx.x;
  for (int i = tid; i < D_; i += 256) hv[i] = head_sum[b * D_ + i];
  __syncthreads();
  for (int c = tid; c < D_; c += 256) {
    float a = final_b[c];
    for (int d = 0; d < D_; ++d) a = fmaf(hv[d], final_w[d * D_ + c], a);
    ov[c] = fmaxf(a, 0.0f);
  }
  __syncthreads();
  if (tid < L_) {
    float a = fc_b[tid];
    for (int d = 0; d < D_; ++d) a = fmaf(ov[d], fc_w[d * L_ + tid], a);
    out[b * L_ + tid] = a;
  }
}

// ---------------------------------------------------------------------------
extern "C" void kernel_launch(void* const* d_in, const int* in_sizes, int n_in,
                              void* d_out, int out_size, void* d_ws, size_t ws_size,
                              hipStream_t stream) {
  const float* word    = (const float*)d_in[0];
  const float* rel     = (const float*)d_in[1];
  const float* mask    = (const float*)d_in[2];
  const float* Wn_w    = (const float*)d_in[3];
  const float* Wn_b    = (const float*)d_in[4];
  const float* Wr_w    = (const float*)d_in[5];
  const float* Wr_b    = (const float*)d_in[6];
  const float* lin_w   = (const float*)d_in[7];
  const float* lin_b   = (const float*)d_in[8];
  const float* score_w = (const float*)d_in[9];
  const float* score_b = (const float*)d_in[10];
  const float* final_w = (const float*)d_in[11];
  const float* final_b = (const float*)d_in[12];
  const float* fc_w    = (const float*)d_in[13];
  const float* fc_b    = (const float*)d_in[14];
  float* out = (float*)d_out;

  char* ws = (char*)d_ws;
  unsigned short* Wb_lin = (unsigned short*)(ws);                   // 512 KB
  unsigned short* Wb_n   = (unsigned short*)(ws +  524288);         // 512 KB
  unsigned short* Wb_r   = (unsigned short*)(ws + 1048576);         // 512 KB
  float* wbuf     = (float*)(ws + 1572864);                         // 4 MB
  float* rbuf     = (float*)(ws + 1572864 + 4194304);               // 2 MB
  float* head_sum = (float*)(ws + 1572864 + 4194304 + 2097152);     // 32 KB

  // Pack weights into bf16 WMMA fragment order; zero the head accumulator.
  pack_lin_kernel<<<dim3(1024), dim3(256), 0, stream>>>(lin_w, Wb_lin, head_sum);
  pack_dd_kernel <<<dim3(1024), dim3(256), 0, stream>>>(Wn_w, Wb_n);
  pack_dd_kernel <<<dim3(1024), dim3(256), 0, stream>>>(Wr_w, Wb_r);

  // Projections via WMMA (64 rows / block -> weight matrix read once/block).
  (void)hipFuncSetAttribute(reinterpret_cast<const void*>(gemm64_kernel),
                            hipFuncAttributeMaxDynamicSharedMemorySize, 65536);
  gemm64_kernel<<<dim3(B_ * N_ / 64), dim3(256), 65536, stream>>>(word, Wb_n, Wn_b, wbuf);
  gemm64_kernel<<<dim3(B_ * M_ / 64), dim3(256), 65536, stream>>>(rel,  Wb_r, Wr_b, rbuf);

  const size_t smem = 65536 + 131072 + 1024 + 2048;  // 199680 B < 320 KB/WGP
  (void)hipFuncSetAttribute(reinterpret_cast<const void*>(main_kernel),
                            hipFuncAttributeMaxDynamicSharedMemorySize,
                            (int)smem);
  main_kernel<<<dim3(B_ * N_), dim3(256), smem, stream>>>(
      wbuf, rbuf, mask, Wb_lin, lin_b, score_w, score_b, head_sum);

  final_kernel<<<dim3(B_), dim3(256), 0, stream>>>(
      head_sum, final_w, final_b, fc_w, fc_b, out);
}
